// AttentionLayer_73306501808959
// MI455X (gfx1250) — matrix operand order repaired and checked by reading, not compiled, not run on hardware
//
#include <hip/hip_runtime.h>
#include <hip/hip_bf16.h>

// Bahdanau additive attention, MI455X (gfx1250, wave32, WMMA).
// Dominant work: keys[B*S,H] @ Ua^T[H,H] = 34.4 GFLOP -> bf16 WMMA, f32 accum.
// keys = 128 MB (L2-resident on 192MB L2); ~2 passes over keys total.

#define B_ 32
#define S_ 2048
#define H_ 512

typedef __attribute__((ext_vector_type(16))) __bf16 v16bf;
typedef __attribute__((ext_vector_type(8)))  float  v8f;

// Native CDNA5 v_tanh_f32 (TRANS op, co-executes with WMMA) when available.
#if __has_builtin(__builtin_amdgcn_tanhf)
  #define TANH_F32(x) __builtin_amdgcn_tanhf(x)
#elif __has_builtin(__builtin_amdgcn_tanh_f32)
  #define TANH_F32(x) __builtin_amdgcn_tanh_f32(x)
#else
  #define TANH_F32(x) tanhf(x)
#endif

// f32 -> bf16 (round-to-nearest-even), stored as raw u16 to keep host code bf16-free.
static __device__ __forceinline__ unsigned short f32_to_bf16u(float f) {
    unsigned u = __builtin_bit_cast(unsigned, f);
    u += 0x7FFFu + ((u >> 16) & 1u);
    return (unsigned short)(u >> 16);
}

// ---------------------------------------------------------------------------
// Prep 1: Ua_w (f32 [H,H], row = output n, col = k) -> fragment-swizzled bf16
// B-matrix fragments for v_wmma_f32_16x16x32_bf16 (B is 32x16 K x N):
//   frag f = nChunk*16 + kStep ; per lane L: N = n0 + (L&15), K = k0 + (L>>4)*8 + (j&7) + (j>>3)*16
//   -> each lane's 16 halves are contiguous (32B) in Bfrag.
// ---------------------------------------------------------------------------
__global__ void __launch_bounds__(256) prep_bfrag(const float* __restrict__ Ua_w,
                                                  unsigned short* __restrict__ Bfrag) {
    int idx = blockIdx.x * 256 + threadIdx.x;     // 0 .. 512*512-1
    int f    = idx >> 9;                          // fragment id
    int lr   = idx & 511;
    int lane = lr >> 4;
    int j    = lr & 15;
    int nChunk = f >> 4, kStep = f & 15;
    int n = nChunk * 16 + (lane & 15);
    int k = kStep * 32 + ((lane >> 4) & 1) * 8 + (j & 7) + (j >> 3) * 16;
    Bfrag[idx] = f32_to_bf16u(Ua_w[n * H_ + k]);  // B[k][n] = Ua_w[n][k]
}

// ---------------------------------------------------------------------------
// Prep 2: qp[b,n] = sum_h query[b,h]*Wa_w[n,h] + Wa_b[n] + Ua_b[n]
// (Va_b omitted everywhere: constant shift of scores is softmax-invariant.)
// ---------------------------------------------------------------------------
__global__ void __launch_bounds__(256) prep_qproj(const float* __restrict__ query,
                                                  const float* __restrict__ Wa_w,
                                                  const float* __restrict__ Wa_b,
                                                  const float* __restrict__ Ua_b,
                                                  float* __restrict__ qp) {
    int idx = blockIdx.x * 256 + threadIdx.x;     // 0 .. 32*512-1
    int b = idx >> 9, o = idx & 511;
    const float4* q4 = (const float4*)(query + (size_t)b * H_);
    const float4* w4 = (const float4*)(Wa_w + (size_t)o * H_);
    float acc = Wa_b[o] + Ua_b[o];
    for (int h = 0; h < H_ / 4; ++h) {
        float4 a = q4[h], c = w4[h];
        acc += a.x * c.x + a.y * c.y + a.z * c.z + a.w * c.w;
    }
    qp[idx] = acc;
}

// ---------------------------------------------------------------------------
// Main: per WG of 256 threads (8 wave32): 64 rows of keys.
//   Stage 64x512 bf16 A-tile in LDS in WMMA-A fragment order (64 KB).
//   A-matrix 16x32 bf16 lane layout (ISA 7.12.2): lane(0-15)=row M, halves j:
//     lanes 0-15:  K = {k0..k0+7, k0+16..k0+23}
//     lanes 16-31: K = {k0+8..k0+15, k0+24..k0+31}
//     => K = k0 + laneHi*8 + (j&7) + (j>>3)*16
//   Wave w: M-subtile (w&3), N-chunks congruent to (w>>2) mod 2 (16 chunks).
//   Fused epilogue: partial[m] += Va[n]*v_tanh(C[m,n] + qp[b,n]); reduce via
//   shfl_xor within 16-lane halves + ds_add_f32 into LDS (aliased over A-tile).
// ---------------------------------------------------------------------------
__global__ void __launch_bounds__(256) scores_kernel(const float* __restrict__ keys,
                                                     const unsigned short* __restrict__ Bfrag,
                                                     const float* __restrict__ qp,
                                                     const float* __restrict__ Va_w,
                                                     float* __restrict__ scores) {
    __shared__ __align__(32) unsigned short aT[64 * 512];   // exactly 64 KB
    const int t    = threadIdx.x;
    const int lane = t & 31;
    const int w    = t >> 5;
    const int rowBase = blockIdx.x * 64;      // 64 | 2048 -> single batch per WG
    const int b = rowBase / S_;

    // ---- stage A tile: 8192 float4 loads, scatter-store bf16x4 (8B) to LDS
    for (int it = 0; it < 32; ++it) {
        int vid = t + it * 256;               // float4 index, 0..8191
        int row_local = vid >> 7;             // 128 float4 per row
        int k4 = (vid & 127) << 2;            // k, multiple of 4
        float4 v = *(const float4*)(keys + (size_t)(rowBase + row_local) * H_ + k4);
        int m_sub = row_local >> 4, m = row_local & 15;
        int kStep = k4 >> 5, ko = k4 & 31;
        int laneHi = (ko >> 3) & 1;
        int j0 = (ko & 7) + ((ko >> 4) & 1) * 8;      // 4-aligned within a run
        int off = ((m_sub * 16 + kStep) * 32 + (m + laneHi * 16)) * 16 + j0;
        unsigned p0 = (unsigned)f32_to_bf16u(v.x) | ((unsigned)f32_to_bf16u(v.y) << 16);
        unsigned p1 = (unsigned)f32_to_bf16u(v.z) | ((unsigned)f32_to_bf16u(v.w) << 16);
        *(uint2*)(&aT[off]) = make_uint2(p0, p1);     // off*2 is 8B aligned
    }
    __syncthreads();

    // ---- WMMA loop
    const int m_sub = w & 3;
    const int ncPar = w >> 2;                 // 0 or 1
    float partial[8] = {0.f, 0.f, 0.f, 0.f, 0.f, 0.f, 0.f, 0.f};
    const unsigned short* apBase = aT + ((size_t)(m_sub * 16) * 512 + lane * 16);
    for (int i = 0; i < 16; ++i) {
        int nChunk = ncPar + i * 2;
        v8f c = {};
        const unsigned short* bp = Bfrag + ((size_t)(nChunk * 16) * 512 + lane * 16);
        #pragma unroll 4
        for (int kStep = 0; kStep < 16; ++kStep) {
            v16bf a  = *(const v16bf*)(apBase + kStep * 512);
            v16bf bm = *(const v16bf*)(bp + kStep * 512);
            c = __builtin_amdgcn_wmma_f32_16x16x32_bf16(false, a, false, bm,
                                                        (short)0, c, false, false);
        }
        int n = nChunk * 16 + (lane & 15);
        float van = Va_w[n];
        float qpn = qp[b * H_ + n];
        #pragma unroll
        for (int r = 0; r < 8; ++r)
            partial[r] += van * TANH_F32(c[r] + qpn);
    }

    // ---- reduce to per-row scores; alias LDS (all A-tile reads done)
    __syncthreads();
    float* sScore = (float*)aT;
    if (t < 64) sScore[t] = 0.0f;
    __syncthreads();
    #pragma unroll
    for (int r = 0; r < 8; ++r) {
        float v = partial[r];
        v += __shfl_xor(v, 8, 32);            // xor<16 stays within each half
        v += __shfl_xor(v, 4, 32);
        v += __shfl_xor(v, 2, 32);
        v += __shfl_xor(v, 1, 32);
        if ((lane & 15) == 0) {
            // lanes 0-15 carry rows M=r, lanes 16-31 carry rows M=r+8
            int row = m_sub * 16 + r + ((lane >> 4) << 3);
            atomicAdd(&sScore[row], v);       // ds_add_f32
        }
    }
    __syncthreads();
    if (t < 64) scores[(size_t)rowBase + t] = sScore[t];
}

// ---------------------------------------------------------------------------
// Softmax over S, in place. One WG per batch row.
// ---------------------------------------------------------------------------
__global__ void __launch_bounds__(256) softmax_kernel(float* __restrict__ scores) {
    __shared__ float red[256];
    int b = blockIdx.x, t = threadIdx.x;
    float* row = scores + (size_t)b * S_;
    float vals[8], lmax = -INFINITY;
    #pragma unroll
    for (int i = 0; i < 8; ++i) { vals[i] = row[t + i * 256]; lmax = fmaxf(lmax, vals[i]); }
    red[t] = lmax; __syncthreads();
    for (int s = 128; s > 0; s >>= 1) { if (t < s) red[t] = fmaxf(red[t], red[t + s]); __syncthreads(); }
    float m = red[0]; __syncthreads();
    float lsum = 0.f;
    #pragma unroll
    for (int i = 0; i < 8; ++i) { vals[i] = __expf(vals[i] - m); lsum += vals[i]; }
    red[t] = lsum; __syncthreads();
    for (int s = 128; s > 0; s >>= 1) { if (t < s) red[t] += red[t + s]; __syncthreads(); }
    float inv = 1.0f / red[0];
    #pragma unroll
    for (int i = 0; i < 8; ++i) row[t + i * 256] = vals[i] * inv;
}

// ---------------------------------------------------------------------------
// out[b,h] = sum_s w[b,s] * keys[b,s,h]; weights staged in LDS chunks.
// ---------------------------------------------------------------------------
__global__ void __launch_bounds__(256) out_kernel(const float* __restrict__ keys,
                                                  const float* __restrict__ wts,
                                                  float* __restrict__ out) {
    __shared__ float lw[256];
    int b = blockIdx.x >> 1;
    int h = ((blockIdx.x & 1) << 8) + threadIdx.x;
    const float* kb = keys + (size_t)b * S_ * H_;
    const float* wb = wts + (size_t)b * S_;
    float acc = 0.f;
    for (int sc = 0; sc < S_; sc += 256) {
        __syncthreads();
        lw[threadIdx.x] = wb[sc + threadIdx.x];
        __syncthreads();
        for (int j = 0; j < 256; ++j)
            acc += lw[j] * kb[(size_t)(sc + j) * H_ + h];
    }
    out[(size_t)b * H_ + h] = acc;
}

// ---------------------------------------------------------------------------
extern "C" void kernel_launch(void* const* d_in, const int* in_sizes, int n_in,
                              void* d_out, int out_size, void* d_ws, size_t ws_size,
                              hipStream_t stream) {
    const float* query = (const float*)d_in[0];   // [B,1,H]
    const float* keys  = (const float*)d_in[1];   // [B,S,H]
    const float* Wa_w  = (const float*)d_in[2];   // [H,H]
    const float* Wa_b  = (const float*)d_in[3];   // [H]
    const float* Ua_w  = (const float*)d_in[4];   // [H,H]
    const float* Ua_b  = (const float*)d_in[5];   // [H]
    const float* Va_w  = (const float*)d_in[6];   // [1,H]
    // d_in[7] = Va_b: softmax-invariant constant, intentionally unused.
    (void)in_sizes; (void)n_in; (void)out_size; (void)ws_size;
    float* out = (float*)d_out;

    char* ws = (char*)d_ws;
    unsigned short* Bfrag = (unsigned short*)ws;                    // 512 KB bf16
    float* qp     = (float*)(ws + (512u << 10));                    //  64 KB
    float* scores = (float*)(ws + (512u << 10) + (64u << 10));      // 256 KB (softmax in place)

    prep_bfrag<<<(H_ * H_) / 256, 256, 0, stream>>>(Ua_w, Bfrag);
    prep_qproj<<<(B_ * H_) / 256, 256, 0, stream>>>(query, Wa_w, Wa_b, Ua_b, qp);
    scores_kernel<<<(B_ * S_) / 64, 256, 0, stream>>>(keys, Bfrag, qp, Va_w, scores);
    softmax_kernel<<<B_, 256, 0, stream>>>(scores);
    out_kernel<<<B_ * 2, 256, 0, stream>>>(keys, scores, out);
}
